// MixtureOfExperts_71090298683749
// MI455X (gfx1250) — matrix operand order repaired and checked by reading, not compiled, not run on hardware
//
#include <hip/hip_runtime.h>

// Problem constants (match reference)
#define NTOK 4096
#define DDIM 512
#define HDIM 2048
#define NEXP 8
#define KTOP 2
#define TM   32            // token-slots per workgroup tile (2 x 16-row WMMA tiles)
#define HC   64            // hidden-dim chunk per iteration
#define NCH  (HDIM / HC)   // 32 chunks
#define CHUNK_ELEMS (HC * DDIM)   // 32768 bf16 elements = 64KB per weight chunk
#define AP   (DDIM + 8)    // Abf row stride (ushorts), keeps rows 16B aligned
#define HP   (HC + 8)      // Hbf row stride (ushorts)

typedef __attribute__((ext_vector_type(16))) __bf16        v16bf;
typedef __attribute__((ext_vector_type(8)))  float         v8f;

#if defined(__HIP_DEVICE_COMPILE__) && __has_builtin(__builtin_amdgcn_tensor_load_to_lds)
#define USE_TDM 1
#else
#define USE_TDM 0
#endif

__device__ __forceinline__ unsigned short f32_bf16(float f) {
  unsigned int u = __builtin_bit_cast(unsigned int, f);
  u += 0x7FFFu + ((u >> 16) & 1u);          // round-to-nearest-even
  return (unsigned short)(u >> 16);
}

// Pack two floats into two bf16 (one VALU op if v_cvt_pk_bf16_f32 exists).
__device__ __forceinline__ unsigned int pk_bf16(float a, float b) {
#if defined(__HIP_DEVICE_COMPILE__) && __has_builtin(__builtin_amdgcn_cvt_pk_bf16_f32)
  return __builtin_bit_cast(unsigned int, __builtin_amdgcn_cvt_pk_bf16_f32(a, b));
#else
  return (unsigned int)f32_bf16(a) | ((unsigned int)f32_bf16(b) << 16);
#endif
}

// Pack 4 floats -> 4 bf16 and store as one 8-byte write.
__device__ __forceinline__ void st4(unsigned short* dst, float a, float b,
                                    float c, float d) {
  uint2 v;
  v.x = pk_bf16(a, b);
  v.y = pk_bf16(c, d);
  *(uint2*)dst = v;
}

__device__ __forceinline__ float gelu_tanh(float x) {
  float x3 = x * x * x;
  return 0.5f * x * (1.0f + tanhf(0.7978845608028654f * (x + 0.044715f * x3)));
}

// A fragment (16x32 bf16) from row-major LDS tile (stride in ushorts).
// Contiguous k order used here (B swizzle matches): lane 0-15 M=lane, elems = K [0,16); lanes 16-31: K [16,32).
// Lowers to two ds_load_b128 (runs contiguous & 16B aligned).
__device__ __forceinline__ v16bf frag_A(const unsigned short* base, int stride,
                                        int kOff, int lane) {
  const int half = lane >> 4;
  const unsigned short* row = base + (lane & 15) * stride + kOff + 16 * half;
  typedef __attribute__((ext_vector_type(8))) unsigned int v8u;
  v8u p;
#pragma unroll
  for (int v = 0; v < 8; ++v) {
    int kb = 2 * v;
    p[v] = (unsigned int)row[kb] | ((unsigned int)row[kb + 1] << 16);
  }
  return __builtin_bit_cast(v16bf, p);
}

// B fragment from pre-swizzled LDS: fragment f = 32 lanes x 16 contiguous bf16
// in exact HW register order -> one aligned 32B read (two ds_load_b128).
__device__ __forceinline__ v16bf frag_B_sw(const unsigned short* Bs, int frag,
                                           int lane) {
  return *(const v16bf*)(Bs + frag * 512 + lane * 16);
}

// Store a 4(K) x 4(N) block of a B tile into fragment-swizzled memory (LDS or global).
// Fragment id = (k>>5)*nFrags + (n>>4); lane = ((k>>4)&1)*16 + (n&15); elem = k&15.
__device__ __forceinline__ void stB4x4(unsigned short* Bs, int nFrags, int k0, int c0,
                                       const float4& r0, const float4& r1,
                                       const float4& r2, const float4& r3) {
  const int fBase    = (k0 >> 5) * nFrags;
  const int laneHalf = ((k0 >> 4) & 1) * 16;
  const int elem     = k0 & 15;
#define STB_COL(cc, f0, f1, f2, f3)                                            \
  {                                                                            \
    int c = c0 + (cc);                                                         \
    unsigned short* dst =                                                      \
        Bs + (fBase + (c >> 4)) * 512 + (laneHalf + (c & 15)) * 16 + elem;     \
    st4(dst, f0, f1, f2, f3);                                                  \
  }
  STB_COL(0, r0.x, r1.x, r2.x, r3.x)
  STB_COL(1, r0.y, r1.y, r2.y, r3.y)
  STB_COL(2, r0.z, r1.z, r2.z, r3.z)
  STB_COL(3, r0.w, r1.w, r2.w, r3.w)
#undef STB_COL
}

#if USE_TDM
typedef unsigned int t_v4u __attribute__((ext_vector_type(4)));
typedef int          t_v8i __attribute__((ext_vector_type(8)));
typedef int          t_v4i __attribute__((ext_vector_type(4)));

// 1-D TDM copy: n2 2-byte elements, contiguous global -> contiguous LDS.
// D# per CDNA5 ISA 8.3/8.4: group0 = {count|flags, lds_addr, global_addr, type=2},
// group1 = {data_size=2B, tensor_dim0=n2, tensor_dim1=1, tile_dim0=n2, tile_dim1=1,
//           tensor_dim0_stride=n2}. This toolchain's builtin takes 6 args
// (g0 v4u, g1 v8i, g2 v4i, g3 v4i, extra v8i, cpol).
__device__ __forceinline__ void tdm_load_lds(unsigned lds_addr, const void* gsrc,
                                             unsigned n2) {
  unsigned long long ga = (unsigned long long)(size_t)gsrc;
  t_v4u g0;
  g0.x = 1u;                                                  // count=1, user D#
  g0.y = lds_addr;                                            // LDS byte address
  g0.z = (unsigned)ga;                                        // global_addr[31:0]
  g0.w = (unsigned)((ga >> 32) & 0x1FFFFFFu) | (2u << 30);    // addr[56:32] | type=2
  t_v8i g1;
  g1[0] = (int)(1u << 16);                                    // data_size=1 -> 2 bytes
  g1[1] = (int)((n2 & 0xFFFFu) << 16);                        // tensor_dim0 lo
  g1[2] = (int)(((n2 >> 16) & 0xFFFFu) | (1u << 16));         // tensor_dim0 hi | dim1 lo=1
  g1[3] = (int)((n2 & 0xFFFFu) << 16);                        // tile_dim0 = n2
  g1[4] = 1;                                                  // tile_dim1 = 1
  g1[5] = (int)n2;                                            // tensor_dim0_stride lo
  g1[6] = 0;
  g1[7] = 0;
  t_v4i gz4 = {0, 0, 0, 0};
  t_v8i gz8 = {0, 0, 0, 0, 0, 0, 0, 0};
  __builtin_amdgcn_tensor_load_to_lds(g0, g1, gz4, gz4, gz8, 0);
}

__device__ __forceinline__ void wait_tensorcnt_le2() {
#if __has_builtin(__builtin_amdgcn_s_wait_tensorcnt)
  __builtin_amdgcn_s_wait_tensorcnt(2);
#else
  asm volatile("s_wait_tensorcnt 0x2" ::: "memory");
#endif
}
__device__ __forceinline__ void wait_tensorcnt_le0() {
#if __has_builtin(__builtin_amdgcn_s_wait_tensorcnt)
  __builtin_amdgcn_s_wait_tensorcnt(0);
#else
  asm volatile("s_wait_tensorcnt 0x0" ::: "memory");
#endif
}
#endif  // USE_TDM

// ---------------- weight prep: fp32 -> bf16, fragment-swizzled chunk blocks ----
// W1bf layout: [e][chunk][frag 0..63][lane 0..31][elem 0..15], 64KB per chunk.
__global__ void moe_prep_w1(const float* __restrict__ W1,
                            unsigned short* __restrict__ W1bf) {
  int idx = blockIdx.x * blockDim.x + threadIdx.x;   // 8 * 128 * 512 4x4 blocks
  int e  = idx >> 16;
  int r  = idx & 65535;
  int h0 = (r & 511) * 4;
  int d0 = (r >> 9) * 4;
  const float* g = W1 + ((size_t)e * DDIM + d0) * HDIM + h0;
  float4 r0 = *(const float4*)(g);
  float4 r1 = *(const float4*)(g + HDIM);
  float4 r2 = *(const float4*)(g + 2 * HDIM);
  float4 r3 = *(const float4*)(g + 3 * HDIM);
  int ch = h0 >> 6, c0 = h0 & 63;
  unsigned short* base = W1bf + (size_t)(e * NCH + ch) * CHUNK_ELEMS;
  stB4x4(base, HC / 16, d0, c0, r0, r1, r2, r3);
}

__global__ void moe_prep_w2(const float* __restrict__ W2,
                            unsigned short* __restrict__ W2bf) {
  int idx = blockIdx.x * blockDim.x + threadIdx.x;   // 8 * 512 * 128 4x4 blocks
  int e  = idx >> 16;
  int r  = idx & 65535;
  int d0 = (r & 127) * 4;
  int h0 = (r >> 7) * 4;
  const float* g = W2 + ((size_t)e * HDIM + h0) * DDIM + d0;
  float4 r0 = *(const float4*)(g);
  float4 r1 = *(const float4*)(g + DDIM);
  float4 r2 = *(const float4*)(g + 2 * DDIM);
  float4 r3 = *(const float4*)(g + 3 * DDIM);
  int ch = h0 >> 6, k0 = h0 & 63;
  unsigned short* base = W2bf + (size_t)(e * NCH + ch) * CHUNK_ELEMS;
  stB4x4(base, DDIM / 16, k0, d0, r0, r1, r2, r3);
}

// ---------------- routing ----------------
__global__ void moe_route_init(int* counts) {
  if (threadIdx.x < NEXP) counts[threadIdx.x] = 0;
}

__global__ void moe_route(const int* __restrict__ idx, int* counts, int* lists) {
  int s = blockIdx.x * blockDim.x + threadIdx.x;   // slot = t*K + k
  if (s < NTOK * KTOP) {
    int e = idx[s];
    int pos = atomicAdd(&counts[e], 1);
    lists[e * NTOK + pos] = s;
  }
}

// ---------------- fused expert MLP ----------------
__global__ __launch_bounds__(256, 1) void moe_expert_kernel(
    const float* __restrict__ X,  const float* __restrict__ probs,
    const unsigned short* __restrict__ W1bf, const float* __restrict__ b1,
    const unsigned short* __restrict__ W2bf, const float* __restrict__ b2,
    const int* __restrict__ counts, const int* __restrict__ lists,
    float* __restrict__ contrib) {
  const int e    = blockIdx.y;
  const int tile = blockIdx.x;
  const int cnt  = counts[e];
  if (tile * TM >= cnt) return;

  __shared__ alignas(32) unsigned short Abf[TM * AP];           // 32 x 520 activations
  __shared__ alignas(32) unsigned short W1s[2][CHUNK_ELEMS];    // double-buffered 64KB
  __shared__ alignas(32) unsigned short W2s[2][CHUNK_ELEMS];    // double-buffered 64KB
  __shared__ alignas(32) unsigned short Hbf[TM * HP];           // 32 x 72 hidden tile
  __shared__ int slotId[TM];

  const int tid  = threadIdx.x;
  const int lane = tid & 31;
  const int wave = tid >> 5;           // 0..7 (wave32)
  const int half = lane >> 4;
  const int n16  = lane & 15;

  const unsigned short* W1e = W1bf + (size_t)e * NCH * CHUNK_ELEMS;
  const unsigned short* W2e = W2bf + (size_t)e * NCH * CHUNK_ELEMS;

  if (tid < TM) {
    int i = tile * TM + tid;
    slotId[tid] = (i < cnt) ? lists[e * NTOK + i] : -1;
  }
  __syncthreads();

  // Gather token rows (fp32 -> bf16) into LDS once (row-major A layout).
  {
    const int c4 = (tid & 127) * 4;
#pragma unroll
    for (int it = 0; it < 16; ++it) {
      int r = (tid >> 7) + 2 * it;
      int s = slotId[r];
      float4 v = make_float4(0.f, 0.f, 0.f, 0.f);
      if (s >= 0) v = *(const float4*)(X + (size_t)(s / KTOP) * DDIM + c4);
      st4(Abf + r * AP + c4, v.x, v.y, v.z, v.w);
    }
  }

#if USE_TDM
  // Prologue: async-stage chunk 0 via the Tensor Data Mover (wave 0 issues).
  if (wave == 0) {
    tdm_load_lds((unsigned)(size_t)(void*)&W1s[0][0], W1e, CHUNK_ELEMS);
    tdm_load_lds((unsigned)(size_t)(void*)&W2s[0][0], W2e, CHUNK_ELEMS);
  }
#endif

  const v8f vzero = {0.f, 0.f, 0.f, 0.f, 0.f, 0.f, 0.f, 0.f};
  v8f acc[2][4];                       // wave's 32 x 64 output slice
#pragma unroll
  for (int mt = 0; mt < 2; ++mt)
#pragma unroll
    for (int q = 0; q < 4; ++q) acc[mt][q] = vzero;

  for (int ch = 0; ch < NCH; ++ch) {
    const int b = ch & 1;
    __syncthreads();  // everyone done reading the buffer TDM is about to overwrite

#if USE_TDM
    if (wave == 0) {
      if (ch + 1 < NCH) {
        tdm_load_lds((unsigned)(size_t)(void*)&W1s[b ^ 1][0],
                     W1e + (size_t)(ch + 1) * CHUNK_ELEMS, CHUNK_ELEMS);
        tdm_load_lds((unsigned)(size_t)(void*)&W2s[b ^ 1][0],
                     W2e + (size_t)(ch + 1) * CHUNK_ELEMS, CHUNK_ELEMS);
        wait_tensorcnt_le2();   // chunk ch's pair retired (TENSORcnt is in-order)
      } else {
        wait_tensorcnt_le0();
      }
    }
#else
    {
      // Fallback: cooperative contiguous b128 copy of current chunk.
      const uint4* s1 = (const uint4*)(W1e + (size_t)ch * CHUNK_ELEMS);
      const uint4* s2 = (const uint4*)(W2e + (size_t)ch * CHUNK_ELEMS);
      uint4* d1 = (uint4*)&W1s[b][0];
      uint4* d2 = (uint4*)&W2s[b][0];
#pragma unroll
      for (int i = 0; i < CHUNK_ELEMS / 8 / 256; ++i) {
        int o = tid + 256 * i;
        d1[o] = s1[o];
        d2[o] = s2[o];
      }
    }
#endif
    __syncthreads();

    const int hc = ch * HC;

    // GEMM1 (all 8 waves): Htile(32x64) = Abf(32x512) @ W1chunk(512x64), +b1, GELU.
    {
      const int mt1 = wave >> 2;
      const int nw  = wave & 3;
      v8f c1 = vzero;
#pragma unroll
      for (int kk = 0; kk < DDIM / 32; ++kk) {
        v16bf a = frag_A(Abf + mt1 * 16 * AP, AP, kk * 32, lane);
        v16bf bfr = frag_B_sw(&W1s[b][0], kk * 4 + nw, lane);
        c1 = __builtin_amdgcn_wmma_f32_16x16x32_bf16(false, a, false, bfr,
                                                     (short)0, c1, false, false);
      }
      const int col  = 16 * nw + n16;
      const float bias = b1[(size_t)e * HDIM + hc + col];
#pragma unroll
      for (int j = 0; j < 8; ++j) {
        int m = mt1 * 16 + j + 8 * half;
        Hbf[m * HP + col] = f32_bf16(gelu_tanh(c1[j] + bias));
      }
    }
    __syncthreads();

    // GEMM2 (all 8 waves): acc(32x64) += Hbf(32x64) @ W2chunk(64x[64*wave..+64)).
#pragma unroll
    for (int mt = 0; mt < 2; ++mt) {
#pragma unroll
      for (int ks = 0; ks < 2; ++ks) {
        v16bf a = frag_A(Hbf + mt * 16 * HP, HP, ks * 32, lane);
#pragma unroll
        for (int q = 0; q < 4; ++q) {
          v16bf bfr = frag_B_sw(&W2s[b][0], ks * 32 + 4 * wave + q, lane);
          acc[mt][q] = __builtin_amdgcn_wmma_f32_16x16x32_bf16(
              false, a, false, bfr, (short)0, acc[mt][q], false, false);
        }
      }
    }
  }

  // Prob-weighted per-slot contributions (unique address per slot -> deterministic).
#pragma unroll
  for (int mt = 0; mt < 2; ++mt) {
#pragma unroll
    for (int q = 0; q < 4; ++q) {
      int col = 64 * wave + 16 * q + n16;
      float bias2 = b2[(size_t)e * DDIM + col];
#pragma unroll
      for (int j = 0; j < 8; ++j) {
        int s = slotId[mt * 16 + j + 8 * half];
        if (s >= 0)
          contrib[(size_t)s * DDIM + col] = probs[s] * (acc[mt][q][j] + bias2);
      }
    }
  }
}

// ---------------- combine (fixed k-order: deterministic) ----------------
__global__ void moe_combine(const float* __restrict__ contrib, float* __restrict__ out) {
  int i = blockIdx.x * blockDim.x + threadIdx.x;   // float4 index over N*D
  if (i < NTOK * DDIM / 4) {
    int t  = i / (DDIM / 4);
    int d4 = i % (DDIM / 4);
    const float4* c0 = (const float4*)contrib + (size_t)(t * KTOP) * (DDIM / 4) + d4;
    float4 a = c0[0], b = c0[DDIM / 4];
    ((float4*)out)[i] = make_float4(a.x + b.x, a.y + b.y, a.z + b.z, a.w + b.w);
  }
}

extern "C" void kernel_launch(void* const* d_in, const int* in_sizes, int n_in,
                              void* d_out, int out_size, void* d_ws, size_t ws_size,
                              hipStream_t stream) {
  const float* X     = (const float*)d_in[0];
  const float* probs = (const float*)d_in[1];
  const int*   idx   = (const int*)d_in[2];
  const float* W1    = (const float*)d_in[3];
  const float* b1    = (const float*)d_in[4];
  const float* W2    = (const float*)d_in[5];
  const float* b2    = (const float*)d_in[6];
  float* out = (float*)d_out;

  char* ws = (char*)d_ws;
  int*   counts  = (int*)ws;                                    // 8 ints (256B pad)
  int*   lists   = (int*)(ws + 256);                            // E*N ints = 128KB
  size_t off     = 256 + (size_t)NEXP * NTOK * sizeof(int);
  float* contrib = (float*)(ws + off);                          // 16MB
  off += (size_t)NTOK * KTOP * DDIM * sizeof(float);
  unsigned short* W1bf = (unsigned short*)(ws + off);           // 16.78MB
  off += (size_t)NEXP * DDIM * HDIM * sizeof(unsigned short);
  unsigned short* W2bf = (unsigned short*)(ws + off);           // 16.78MB

  // One-time (per launch) weight convert + swizzle.
  moe_prep_w1<<<(NEXP * 65536) / 256, 256, 0, stream>>>(W1, W1bf);
  moe_prep_w2<<<(NEXP * 65536) / 256, 256, 0, stream>>>(W2, W2bf);

  moe_route_init<<<1, 32, 0, stream>>>(counts);
  moe_route<<<(NTOK * KTOP + 255) / 256, 256, 0, stream>>>(idx, counts, lists);

  dim3 grid(NTOK / TM, NEXP);  // 128 tiles x 8 experts, early-exit past count
  moe_expert_kernel<<<grid, 256, 0, stream>>>(X, probs, W1bf, b1, W2bf, b2,
                                              counts, lists, contrib);

  moe_combine<<<(NTOK * DDIM / 4 + 255) / 256, 256, 0, stream>>>(contrib, out);
}
